// AnswerGenerator_56169582297257
// MI455X (gfx1250) — matrix operand order repaired and checked by reading, not compiled, not run on hardware
//
// MI455X / gfx1250 transformer decoder.
// Weight-bandwidth-bound => bf16 weights + bf16 activations, all GEMMs via
// v_wmma_f32_16x16x32_bf16 (16x32 tile/wave, 4 wmma per K-chunk), weight
// tiles streamed into LDS by the Tensor Data Mover with DOUBLE BUFFERING
// (one tensor_load_to_lds in flight while computing: s_wait_tensorcnt 1 in
// steady state, 0 on the last tile; D# pad fields insert 16B per 128B line
// for conflict-free ds_load_b128). Attention/LN/softmax on VALU+LDS,
// cross-attn K/V hoisted out of the decode loop.

#include <hip/hip_runtime.h>
#include <hip/hip_bf16.h>
#include <stdint.h>

#define B_     2
#define S0     64
#define D_     1024
#define V_     32000
#define H_     8
#define L_     2
#define HD_    128
#define FF_    4096
#define NSTEPS 8
#define SMAX   72
#define KC     64           // K chunk staged per LDS tile
#define LDB    (KC + 8)     // LDS row stride in elems (144B; 16B pad per 128B line)

typedef __attribute__((ext_vector_type(16))) __bf16         v16bf;
typedef __attribute__((ext_vector_type(8)))  float          v8f;
typedef __attribute__((ext_vector_type(16))) unsigned short v16us;
typedef __attribute__((ext_vector_type(8)))  unsigned short ushort8;
typedef __attribute__((ext_vector_type(4)))  unsigned int   u32x4;
typedef __attribute__((ext_vector_type(8)))  int            i32x8;
typedef __attribute__((ext_vector_type(4)))  int            i32x4;

struct AU { ushort8 lo, hi; };

// ---------------- small device helpers ----------------

__device__ __forceinline__ unsigned short f2bf(float f) {
  union { float f; unsigned u; } x; x.f = f;
  unsigned u = x.u + 0x7FFFu + ((x.u >> 16) & 1u);
  return (unsigned short)(u >> 16);
}
__device__ __forceinline__ float bf2f(unsigned short h) {
  union { unsigned u; float f; } x; x.u = ((unsigned)h) << 16;
  return x.f;
}
__device__ __forceinline__ float wsum(float v) {
#pragma unroll
  for (int off = 16; off > 0; off >>= 1) v += __shfl_xor(v, off, 32);
  return v;
}
__device__ __forceinline__ float wmaxf(float v) {
#pragma unroll
  for (int off = 16; off > 0; off >>= 1) v = fmaxf(v, __shfl_xor(v, off, 32));
  return v;
}
__device__ __forceinline__ float block_sum(float v, float* red) {
  const int lane = threadIdx.x & 31, wid = threadIdx.x >> 5, nw = blockDim.x >> 5;
  v = wsum(v);
  if (lane == 0) red[wid] = v;
  __syncthreads();
  float t = (threadIdx.x < (unsigned)nw) ? red[threadIdx.x] : 0.f;
  if (wid == 0) { t = wsum(t); if (lane == 0) red[0] = t; }
  __syncthreads();
  float r = red[0];
  __syncthreads();
  return r;
}
__device__ __forceinline__ unsigned smix(unsigned x) {
  x ^= x >> 16; x *= 0x7feb352dU; x ^= x >> 15; x *= 0x846ca68bU; x ^= x >> 16;
  return x;
}

// ---------------- conversion / copy ----------------

__global__ void k_cvt(const float* __restrict__ s, unsigned short* __restrict__ d, long n) {
  long i  = (long)blockIdx.x * blockDim.x + threadIdx.x;
  long st = (long)gridDim.x * blockDim.x;
  for (; i < n; i += st) d[i] = f2bf(s[i]);
}

__global__ void k_copyf(const float* __restrict__ s, float* __restrict__ d, long n) {
  long i  = (long)blockIdx.x * blockDim.x + threadIdx.x;
  long st = (long)gridDim.x * blockDim.x;
  for (; i < n; i += st) d[i] = s[i];
}

__global__ void k_copy_cur(const float* __restrict__ emb, float* __restrict__ cur) {
  long i = (long)blockIdx.x * blockDim.x + threadIdx.x;
  long n = (long)B_ * S0 * D_;
  if (i >= n) return;
  long b = i / ((long)S0 * D_);
  long r = i % ((long)S0 * D_);
  cur[b * (long)SMAX * D_ + r] = emb[i];
}

// ---------------- WMMA GEMM, double-buffered TDM weight streaming ----------
// Y[b] = A[b] @ W^T + bias.  A: bf16 [batch, M, lda].  W: bf16 [N, K].
// Block = 128 thr (4 waves) covers 128 output columns; wave = 16x32 tile
// (2 accumulators sharing one A fragment). TDM streams [128 x KC] weight
// tiles into 2 LDS buffers; tile i+1 is in flight while tile i is consumed.
// Fragment layouts used here (wave32); A/B use contiguous k, not the ISA table order:
//   A 16x32 bf16 : lane L -> row (L&15); K = 16*(L>>4)+{0..15} contiguous.
//   B 32x16 bf16 : lane L -> col (L&15); K = 16*(L>>4)+{0..15} contiguous.
//   C/D 16x16 f32: vgpr r -> row r + 8*(L>>4); col = L&15.

__device__ __forceinline__ v8f wmma_bf16(v16bf a, v16us b, v8f c) {
  return __builtin_amdgcn_wmma_f32_16x16x32_bf16(
      false, a, false, __builtin_bit_cast(v16bf, b), (short)0, c, false, false);
}

__global__ void k_gemm(const unsigned short* __restrict__ A, long sAb, int lda,
                       const unsigned short* __restrict__ W,
                       const float* __restrict__ bias,
                       float* __restrict__ Y, long sYb, int ldy,
                       int M, int N, int K, int relu) {
  const int lane = threadIdx.x & 31;
  const int wid  = threadIdx.x >> 5;
  const int half = lane >> 4;
  const int l16  = lane & 15;
  const int m0   = blockIdx.y * 16;
  const int nblk = blockIdx.x * 128;

  __shared__ __align__(32) unsigned short sB[2][128 * LDB];

  const int arow = m0 + l16;
  const unsigned short* Arow = A + (long)blockIdx.z * sAb + (long)arow * lda;

  const int c0 = wid * 32 + l16;   // local columns for tile 0 / tile 1
  const int c1 = c0 + 16;

  v8f acc0 = {}, acc1 = {};
  const int ntiles = K / KC;

#if __has_builtin(__builtin_amdgcn_tensor_load_to_lds)
  // ---- Tensor Data Mover issue: [128 rows x KC elems] of W -> LDS buf ----
  auto tdm_issue = [&](int tile, int buf) {
    unsigned long long ga =
        (unsigned long long)(const void*)(W + (long)nblk * K + (long)tile * KC);
    unsigned ldso = (unsigned)(unsigned long long)(const void*)&sB[buf][0];
    u32x4 g0;
    g0[0] = 1u;                                               // count=1 (valid D#)
    g0[1] = ldso;                                             // lds_addr
    g0[2] = (unsigned)(ga & 0xffffffffu);                     // global_addr[31:0]
    g0[3] = (unsigned)((ga >> 32) & 0x1ffffffu) | (2u << 30); // addr[56:32], type=2
    i32x8 g1;
    g1[0] = (int)((1u << 16) | (1u << 20) | (4u << 22) | (3u << 25));
    //           data_size=2B | pad_en   | intvl=32dw  | pad=4dw (16B per 128B line)
    g1[1] = (int)(((unsigned)K & 0xffffu) << 16);             // tensor_dim0[15:0]
    g1[2] = (int)((((unsigned)K >> 16) & 0xffffu) | (((unsigned)N & 0xffffu) << 16));
    g1[3] = (int)(((((unsigned)N >> 16) & 0xffffu)) | ((unsigned)KC << 16)); // tile_dim0
    g1[4] = 128;                                              // tile_dim1 = 128 rows
    g1[5] = (int)(unsigned)K;                                 // tensor_dim0_stride lo32
    g1[6] = 0;
    g1[7] = 0;
    i32x4 z4 = {};
#if defined(__clang_major__) && (__clang_major__ >= 23)
    i32x8 z8 = {};
    __builtin_amdgcn_tensor_load_to_lds(g0, g1, z4, z4, z8, 0);
#else
    __builtin_amdgcn_tensor_load_to_lds(g0, g1, z4, z4, 0);
#endif
  };
#endif

  auto compute_tile = [&](const unsigned short* sb, int kglob) {
#pragma unroll
    for (int ks = 0; ks < 2; ++ks) {
      const int kk = kglob + ks * 32;
      AU a2; a2.lo = (ushort8){}; a2.hi = (ushort8){};
      if (arow < M) {
        const unsigned short* p0 = Arow + kk + 16 * half;
        a2.lo = *(const ushort8*)p0;
        a2.hi = *(const ushort8*)(p0 + 8);
      }
      v16bf av = __builtin_bit_cast(v16bf, a2);
      v16us bu0 = *(const v16us*)(sb + (long)c0 * LDB + ks * 32 + 16 * half);
      v16us bu1 = *(const v16us*)(sb + (long)c1 * LDB + ks * 32 + 16 * half);
      acc0 = wmma_bf16(av, bu0, acc0);
      acc1 = wmma_bf16(av, bu1, acc1);
    }
  };

#if __has_builtin(__builtin_amdgcn_tensor_load_to_lds)
  // ---- pipelined: keep one TDM in flight while computing ----
  if (wid == 0) {
    tdm_issue(0, 0);
    if (ntiles > 1) tdm_issue(1, 1);
  }
  for (int i = 0; i < ntiles; ++i) {
    if (wid == 0) {
      if (i == ntiles - 1) __builtin_amdgcn_s_wait_tensorcnt(0);  // last tile done
      else                 __builtin_amdgcn_s_wait_tensorcnt(1);  // tile i done (in-order)
    }
    __syncthreads();                      // tile i visible to all waves
    compute_tile(&sB[i & 1][0], i * KC);
    __syncthreads();                      // all readers done with buf (i&1)
    if (wid == 0 && i + 2 < ntiles) tdm_issue(i + 2, i & 1);
  }
#else
  // ---- fallback: cooperative synchronous staging ----
  for (int i = 0; i < ntiles; ++i) {
    __syncthreads();
    {
      const int tid = threadIdx.x;
      const unsigned short* Wr = W + (long)(nblk + tid) * K + i * KC;
#pragma unroll
      for (int j = 0; j < KC / 8; ++j)
        *(ushort8*)(&sB[0][0] + (long)tid * LDB + j * 8) = *(const ushort8*)(Wr + j * 8);
    }
    __syncthreads();
    compute_tile(&sB[0][0], i * KC);
  }
#endif

  float* Yb = Y + (long)blockIdx.z * sYb;
  const int gc0 = nblk + c0, gc1 = nblk + c1;
  const float bv0 = bias[gc0], bv1 = bias[gc1];
#pragma unroll
  for (int r = 0; r < 8; ++r) {
    int m = m0 + 8 * half + r;
    if (m < M) {
      float y0 = acc0[r] + bv0, y1 = acc1[r] + bv1;
      if (relu) { y0 = fmaxf(y0, 0.f); y1 = fmaxf(y1, 0.f); }
      Yb[(long)m * ldy + gc0] = y0;
      Yb[(long)m * ldy + gc1] = y1;
    }
  }
}

// ---------------- attention (one wave per (q, head, batch)) ----------------

__global__ void k_attn(const float* __restrict__ Q, long qbs, int qld,
                       const float* __restrict__ KV, long kbs, int kld, int voff,
                       float* __restrict__ O, long obs, int oldd,
                       int Sk, float scale) {
  const int qi = blockIdx.x, h = blockIdx.y, b = blockIdx.z;
  const int lane = threadIdx.x & 31;
  __shared__ float sc[SMAX];

  const float4 qv = *(const float4*)(Q + (long)b * qbs + (long)qi * qld + h * HD_ + lane * 4);
  const float* kb = KV + (long)b * kbs + h * HD_ + lane * 4;
  for (int j = 0; j < Sk; ++j) {
    float4 kv = *(const float4*)(kb + (long)j * kld);
    float p = qv.x * kv.x + qv.y * kv.y + qv.z * kv.z + qv.w * kv.w;
    p = wsum(p);
    if (lane == 0) sc[j] = p * scale;
  }
  __syncthreads();

  float mx = -1e30f;
  for (int j = lane; j < Sk; j += 32) mx = fmaxf(mx, sc[j]);
  mx = wmaxf(mx);
  float sum = 0.f;
  for (int j = lane; j < Sk; j += 32) { float e = expf(sc[j] - mx); sc[j] = e; sum += e; }
  sum = wsum(sum);
  __syncthreads();
  const float inv = 1.f / sum;

  float4 accv = {0.f, 0.f, 0.f, 0.f};
  const float* vb = KV + (long)b * kbs + voff + h * HD_ + lane * 4;
  for (int j = 0; j < Sk; ++j) {
    float p = sc[j] * inv;
    float4 vv = *(const float4*)(vb + (long)j * kld);
    accv.x += p * vv.x; accv.y += p * vv.y; accv.z += p * vv.z; accv.w += p * vv.w;
  }
  *(float4*)(O + (long)b * obs + (long)qi * oldd + h * HD_ + lane * 4) = accv;
}

// ---------------- LayerNorm(x + r) ----------------

__global__ void k_ln_res(const float* __restrict__ X, long xbs,
                         const float* __restrict__ R, long rbs,
                         const float* __restrict__ w, const float* __restrict__ bb,
                         float* __restrict__ Y, long ybs) {
  const int s = blockIdx.x, b = blockIdx.y;
  __shared__ float red[8];
  const long xo = (long)b * xbs + (long)s * D_;
  const long ro = (long)b * rbs + (long)s * D_;
  float v[4];
#pragma unroll
  for (int i = 0; i < 4; ++i) {
    int d = threadIdx.x + i * 256;
    v[i] = X[xo + d] + R[ro + d];
  }
  float mean = block_sum(v[0] + v[1] + v[2] + v[3], red) * (1.f / D_);
  float lv = 0.f;
#pragma unroll
  for (int i = 0; i < 4; ++i) { float t = v[i] - mean; lv += t * t; }
  float var = block_sum(lv, red) * (1.f / D_);
  float rs = rsqrtf(var + 1e-5f);
  const long yo = (long)b * ybs + (long)s * D_;
#pragma unroll
  for (int i = 0; i < 4; ++i) {
    int d = threadIdx.x + i * 256;
    Y[yo + d] = (v[i] - mean) * rs * w[d] + bb[d];
  }
}

// ---------------- control heads ----------------

__global__ void k_embmean(const float* __restrict__ emb, float* __restrict__ out) {
  int idx = blockIdx.x * blockDim.x + threadIdx.x;
  if (idx >= B_ * D_) return;
  int b = idx / D_, d = idx % D_;
  float s = 0.f;
  for (int t = 0; t < S0; ++t) s += emb[((long)b * S0 + t) * D_ + d];
  out[idx] = s * (1.f / S0);
}

__global__ void k_temp(const float* __restrict__ emb, const float* __restrict__ tw,
                       const float* __restrict__ tb, float* __restrict__ tempv) {
  const int b = blockIdx.x;
  __shared__ float red[8];
  float acc = 0.f;
  for (int s = 0; s < S0; ++s) {
    float part = 0.f;
    for (int d = threadIdx.x; d < D_; d += blockDim.x)
      part += emb[((long)b * S0 + s) * D_ + d] * tw[d];
    float dot = block_sum(part, red);
    acc += 1.f / (1.f + expf(-(dot + tb[0])));
  }
  if (threadIdx.x == 0) tempv[b] = acc / (float)S0;
}

// ---------------- logits (one wave per vocab row, both batches) ----------------

__global__ void k_logits(const float* __restrict__ Hb, long hbs, int St,
                         const float* __restrict__ style,
                         const unsigned short* __restrict__ Wo,
                         const float* __restrict__ bo,
                         const float* __restrict__ tempv,
                         float* __restrict__ lgs, int t,
                         float* __restrict__ lws) {
  const int lane = threadIdx.x & 31;
  const int wid  = threadIdx.x >> 5;
  const int v = blockIdx.x * 8 + wid;
  if (v >= V_) return;
  const unsigned short* wr = Wo + (long)v * D_;
  __builtin_prefetch(wr + 8 * D_, 0, 0);     // global_prefetch_b8 a few rows ahead
  const float* h0 = Hb + (long)(St - 1) * D_;
  const float* h1 = Hb + hbs + (long)(St - 1) * D_;
  float a0 = 0.f, a1 = 0.f;
  for (int i = 0; i < D_ / 32; ++i) {
    int k = i * 32 + lane;
    float w = bf2f(wr[k]);
    a0 += w * (h0[k] + style[k]);
    a1 += w * (h1[k] + style[D_ + k]);
  }
  a0 = wsum(a0); a1 = wsum(a1);
  if (lane == 0) {
    float l0 = (a0 + bo[v]) / tempv[0];
    float l1 = (a1 + bo[v]) / tempv[1];
    lws[v] = l0; lws[V_ + v] = l1;
    lgs[(long)(0 * NSTEPS + t) * V_ + v] = l0;
    lgs[(long)(1 * NSTEPS + t) * V_ + v] = l1;
  }
}

// ---------------- deterministic Gumbel-argmax sampling ----------------

__global__ void k_sample(const float* __restrict__ lws, int t,
                         int* __restrict__ tok, float* __restrict__ toks_out) {
  const int b = blockIdx.x;
  __shared__ float bv[256];
  __shared__ int   bi[256];
  float best = -1e30f; int besti = 0;
  for (int v = threadIdx.x; v < V_; v += 256) {
    unsigned hsh = smix(0x9E3779B9u * (unsigned)(t + 1) ^
                        0x85EBCA6Bu * (unsigned)(b + 1) ^ (unsigned)v);
    float u = ((hsh >> 8) + 0.5f) * (1.f / 16777216.f);
    float g = -logf(-logf(u));
    float val = lws[(long)b * V_ + v] + g;
    if (val > best) { best = val; besti = v; }
  }
  bv[threadIdx.x] = best; bi[threadIdx.x] = besti;
  __syncthreads();
  for (int s2 = 128; s2 > 0; s2 >>= 1) {
    if (threadIdx.x < (unsigned)s2 && bv[threadIdx.x + s2] > bv[threadIdx.x]) {
      bv[threadIdx.x] = bv[threadIdx.x + s2]; bi[threadIdx.x] = bi[threadIdx.x + s2];
    }
    __syncthreads();
  }
  if (threadIdx.x == 0) {
    tok[b] = bi[0];
    toks_out[(long)b * NSTEPS + t] = (float)bi[0];
  }
}

__global__ void k_append(const float* __restrict__ outw, const int* __restrict__ tok,
                         float* __restrict__ cur, int St) {
  int d = blockIdx.x * blockDim.x + threadIdx.x;
  int b = blockIdx.y;
  if (d < D_) cur[((long)b * SMAX + St) * D_ + d] = outw[(long)tok[b] * D_ + d];
}

// ---------------- host orchestration ----------------

extern "C" void kernel_launch(void* const* d_in, const int* in_sizes, int n_in,
                              void* d_out, int out_size, void* d_ws, size_t ws_size,
                              hipStream_t stream) {
  (void)in_sizes; (void)n_in; (void)out_size; (void)ws_size;

  const float* emb      = (const float*)d_in[0];
  const float* sa_in_w  = (const float*)d_in[1];
  const float* sa_in_b  = (const float*)d_in[2];
  const float* sa_out_w = (const float*)d_in[3];
  const float* sa_out_b = (const float*)d_in[4];
  const float* ca_in_w  = (const float*)d_in[5];
  const float* ca_in_b  = (const float*)d_in[6];
  const float* ca_out_w = (const float*)d_in[7];
  const float* ca_out_b = (const float*)d_in[8];
  const float* ff1_w    = (const float*)d_in[9];
  const float* ff1_b    = (const float*)d_in[10];
  const float* ff2_w    = (const float*)d_in[11];
  const float* ff2_b    = (const float*)d_in[12];
  const float* ln_w     = (const float*)d_in[13];
  const float* ln_b     = (const float*)d_in[14];
  const float* out_w    = (const float*)d_in[15];
  const float* out_b    = (const float*)d_in[16];
  const float* temp_w   = (const float*)d_in[17];
  const float* temp_b   = (const float*)d_in[18];
  const float* style_w  = (const float*)d_in[21];
  const float* style_b  = (const float*)d_in[22];

  char* base = (char*)d_ws;
  size_t off = 0;
  auto alloc = [&](size_t bytes) -> void* {
    off = (off + 255) & ~(size_t)255;
    void* p = base + off; off += bytes; return p;
  };
  auto abf  = [&](size_t n) { return (unsigned short*)alloc(n * 2); };
  auto af32 = [&](size_t n) { return (float*)alloc(n * 4); };

  unsigned short* sa_in_bf  = abf((size_t)L_ * 3 * D_ * D_);
  unsigned short* sa_out_bf = abf((size_t)L_ * D_ * D_);
  unsigned short* ca_in_bf  = abf((size_t)L_ * 3 * D_ * D_);
  unsigned short* ca_out_bf = abf((size_t)L_ * D_ * D_);
  unsigned short* ff1_bf    = abf((size_t)L_ * FF_ * D_);
  unsigned short* ff2_bf    = abf((size_t)L_ * D_ * FF_);
  unsigned short* out_bf    = abf((size_t)V_ * D_);
  unsigned short* style_bf  = abf((size_t)D_ * D_);

  unsigned short* emb_bf     = abf((size_t)B_ * S0 * D_);
  unsigned short* embmean_bf = abf((size_t)B_ * D_);
  unsigned short* xb_bf      = abf((size_t)B_ * SMAX * D_);
  unsigned short* ob_bf      = abf((size_t)B_ * SMAX * D_);
  unsigned short* ffh_bf     = abf((size_t)B_ * SMAX * FF_);

  float* embmean = af32((size_t)B_ * D_);
  float* tempv   = af32(B_);
  float* stylev  = af32((size_t)B_ * D_);
  float* ca_kv   = af32((size_t)L_ * B_ * S0 * 2 * D_);
  float* cur     = af32((size_t)B_ * SMAX * D_);
  float* xb      = af32((size_t)B_ * SMAX * D_);
  float* qkv     = af32((size_t)B_ * SMAX * 3 * D_);
  float* ob      = af32((size_t)B_ * SMAX * D_);
  float* x2      = af32((size_t)B_ * SMAX * D_);
  float* ffh     = af32((size_t)B_ * SMAX * FF_);
  float* lws     = af32((size_t)B_ * V_);
  int*   tok     = (int*)alloc(B_ * sizeof(int));

  auto cvt = [&](const float* s, unsigned short* d, long n) {
    k_cvt<<<dim3(1024), dim3(256), 0, stream>>>(s, d, n);
  };
  auto gemm = [&](const unsigned short* A, long sAb, int lda, const unsigned short* W,
                  const float* bias, float* Y, long sYb, int ldy,
                  int M, int N, int K, int relu, int batch) {
    dim3 g(N / 128, (M + 15) / 16, batch);
    k_gemm<<<g, dim3(128), 0, stream>>>(A, sAb, lda, W, bias, Y, sYb, ldy, M, N, K, relu);
  };

  // one-time: weights and step-invariant activations -> bf16
  cvt(sa_in_w,  sa_in_bf,  (long)L_ * 3 * D_ * D_);
  cvt(sa_out_w, sa_out_bf, (long)L_ * D_ * D_);
  cvt(ca_in_w,  ca_in_bf,  (long)L_ * 3 * D_ * D_);
  cvt(ca_out_w, ca_out_bf, (long)L_ * D_ * D_);
  cvt(ff1_w,    ff1_bf,    (long)L_ * FF_ * D_);
  cvt(ff2_w,    ff2_bf,    (long)L_ * D_ * FF_);
  cvt(out_w,    out_bf,    (long)V_ * D_);
  cvt(style_w,  style_bf,  (long)D_ * D_);
  cvt(emb,      emb_bf,    (long)B_ * S0 * D_);

  // control heads (step-invariant)
  k_embmean<<<dim3((B_ * D_ + 255) / 256), 256, 0, stream>>>(emb, embmean);
  cvt(embmean, embmean_bf, (long)B_ * D_);
  k_temp<<<dim3(B_), 256, 0, stream>>>(emb, temp_w, temp_b, tempv);
  gemm(embmean_bf, 0, D_, style_bf, style_b, stylev, 0, D_, B_, D_, D_, 0, 1);

  // cross-attention K/V: step-invariant, once per layer
  for (int l = 0; l < L_; ++l)
    gemm(emb_bf, (long)S0 * D_, D_,
         ca_in_bf + (long)l * 3 * D_ * D_ + (long)D_ * D_,   // wk,wv rows [D..3D)
         ca_in_b + l * 3 * D_ + D_,
         ca_kv + (long)l * B_ * S0 * 2 * D_, (long)S0 * 2 * D_, 2 * D_,
         S0, 2 * D_, D_, 0, B_);

  k_copy_cur<<<dim3((B_ * S0 * D_ + 255) / 256), 256, 0, stream>>>(emb, cur);

  const float scale = 0.08838834764831845f;   // 1/sqrt(HD)
  float* lgs_base = (float*)d_out + B_ * NSTEPS;

  for (int t = 0; t < NSTEPS; ++t) {
    const int St = S0 + t;
    k_copyf<<<dim3(1024), 256, 0, stream>>>(cur, xb, (long)B_ * SMAX * D_);

    for (int l = 0; l < L_; ++l) {
      const float* lnw = ln_w + (long)(l * 3) * D_;
      const float* lnb = ln_b + (long)(l * 3) * D_;

      // --- self attention ---
      cvt(xb, xb_bf, (long)B_ * SMAX * D_);
      gemm(xb_bf, (long)SMAX * D_, D_, sa_in_bf + (long)l * 3 * D_ * D_,
           sa_in_b + l * 3 * D_, qkv, (long)SMAX * 3 * D_, 3 * D_,
           St, 3 * D_, D_, 0, B_);
      k_attn<<<dim3(St, H_, B_), 32, 0, stream>>>(
          qkv, (long)SMAX * 3 * D_, 3 * D_,
          qkv + D_, (long)SMAX * 3 * D_, 3 * D_, D_,
          ob, (long)SMAX * D_, D_, St, scale);
      cvt(ob, ob_bf, (long)B_ * SMAX * D_);
      gemm(ob_bf, (long)SMAX * D_, D_, sa_out_bf + (long)l * D_ * D_,
           sa_out_b + l * D_, x2, (long)SMAX * D_, D_, St, D_, D_, 0, B_);
      k_ln_res<<<dim3(St, B_), 256, 0, stream>>>(
          xb, (long)SMAX * D_, x2, (long)SMAX * D_,
          lnw + 0 * D_, lnb + 0 * D_, xb, (long)SMAX * D_);

      // --- cross attention (K/V precomputed) ---
      cvt(xb, xb_bf, (long)B_ * SMAX * D_);
      gemm(xb_bf, (long)SMAX * D_, D_, ca_in_bf + (long)l * 3 * D_ * D_, // wq [0..D)
           ca_in_b + l * 3 * D_, qkv, (long)SMAX * 3 * D_, 3 * D_,
           St, D_, D_, 0, B_);
      k_attn<<<dim3(St, H_, B_), 32, 0, stream>>>(
          qkv, (long)SMAX * 3 * D_, 3 * D_,
          ca_kv + (long)l * B_ * S0 * 2 * D_, (long)S0 * 2 * D_, 2 * D_, D_,
          ob, (long)SMAX * D_, D_, S0, scale);
      cvt(ob, ob_bf, (long)B_ * SMAX * D_);
      gemm(ob_bf, (long)SMAX * D_, D_, ca_out_bf + (long)l * D_ * D_,
           ca_out_b + l * D_, x2, (long)SMAX * D_, D_, St, D_, D_, 0, B_);
      k_ln_res<<<dim3(St, B_), 256, 0, stream>>>(
          xb, (long)SMAX * D_, x2, (long)SMAX * D_,
          lnw + 1 * D_, lnb + 1 * D_, xb, (long)SMAX * D_);

      // --- FFN ---
      cvt(xb, xb_bf, (long)B_ * SMAX * D_);
      gemm(xb_bf, (long)SMAX * D_, D_, ff1_bf + (long)l * FF_ * D_,
           ff1_b + l * FF_, ffh, (long)SMAX * FF_, FF_, St, FF_, D_, 1, B_);
      cvt(ffh, ffh_bf, (long)B_ * SMAX * FF_);
      gemm(ffh_bf, (long)SMAX * FF_, FF_, ff2_bf + (long)l * D_ * FF_,
           ff2_b + l * D_, x2, (long)SMAX * D_, D_, St, D_, FF_, 0, B_);
      k_ln_res<<<dim3(St, B_), 256, 0, stream>>>(
          xb, (long)SMAX * D_, x2, (long)SMAX * D_,
          lnw + 2 * D_, lnb + 2 * D_, xb, (long)SMAX * D_);
    }

    k_logits<<<dim3(V_ / 8), 256, 0, stream>>>(
        xb, (long)SMAX * D_, St, stylev, out_bf, out_b, tempv, lgs_base, t, lws);
    k_sample<<<dim3(B_), 256, 0, stream>>>(lws, t, tok, (float*)d_out);
    k_append<<<dim3((D_ + 255) / 256, B_), 256, 0, stream>>>(out_w, tok, cur, St);
  }
}
